// SparseMoeBlock_85487029059974
// MI455X (gfx1250) — matrix operand order repaired and checked by reading, not compiled, not run on hardware
//
#include <hip/hip_runtime.h>

// ---------------------------------------------------------------------------
// Sparse MoE block for MI455X (gfx1250, wave32, WMMA bf16)
//   T=2048 tokens, H=512, E=16 experts, top-2, I=2048 routed, IS=1024 shared
// Strategy: gate+top2 -> per-expert token lists; 32-token tiles per block;
// bf16 WMMA with f32 accumulate; weights converted f32->bf16 in-register via
// v_perm packing; X tile staged with async global->LDS; shared expert writes
// out, routed experts atomically add.
// ---------------------------------------------------------------------------

#define HDIM  512
#define TOK   2048
#define NEXP  16
#define MTILE 32                 // tokens per block
#define MAXT  (TOK / MTILE)      // max tiles per expert (64)

typedef unsigned short ushort_t;
typedef __attribute__((ext_vector_type(16))) __bf16 v16bf;
typedef __attribute__((ext_vector_type(8)))  float  v8f;

// float -> bf16 bits, round-to-nearest-even (scalar path, epilogues only)
__device__ __forceinline__ ushort_t f2bfu(float f) {
    union { float f; unsigned u; } v; v.f = f;
    unsigned r = v.u + 0x7fffu + ((v.u >> 16) & 1u);
    return (ushort_t)(r >> 16);
}

// float bits + round-half-up for the packed v_perm bf16 path
__device__ __forceinline__ unsigned bfr(float f) {
    union { float f; unsigned u; } v; v.f = f;
    return v.u + 0x8000u;
}

__device__ __forceinline__ v8f wmma_bf16(v16bf a, v16bf b, v8f c) {
    return __builtin_amdgcn_wmma_f32_16x16x32_bf16(
        false, a, false, b, (short)0, c, false, false);
}

// B fragment: two 8-fp32 runs (p, p+16) of one weight row (the lane's K-slice,
// 16B aligned) -> v16bf. One round-add per element + one v_perm per pair;
// fragment assembled as 8 packed dwords (no sub-word moves).
__device__ __forceinline__ v16bf load_b16(const float* __restrict__ p) {
    const float4* q = (const float4*)p;
    unsigned pk[8];
#pragma unroll
    for (int i = 0; i < 4; ++i) {
        float4 f = q[i + (i >> 1) * 2];
        pk[2 * i + 0] = __builtin_amdgcn_perm(bfr(f.y), bfr(f.x), 0x07060302u);
        pk[2 * i + 1] = __builtin_amdgcn_perm(bfr(f.w), bfr(f.z), 0x07060302u);
    }
    v16bf b;
    __builtin_memcpy(&b, pk, 32);
    return b;
}

// A fragment: two 16B-aligned 8-element bf16 runs from LDS (ds_load_b128 x2).
// Lane layout per CDNA5 ISA 7.12.2: e<8 -> K = kb+8*half+e,
//                                   e>=8 -> K = kb+16+8*half+(e-8).
__device__ __forceinline__ v16bf load_a16(const ushort_t* p) {
    uint4 lo = *(const uint4*)(p);
    uint4 hi = *(const uint4*)(p + 16);
    unsigned pk[8] = {lo.x, lo.y, lo.z, lo.w, hi.x, hi.y, hi.z, hi.w};
    v16bf a;
    __builtin_memcpy(&a, pk, 32);
    return a;
}

// ---------------------------------------------------------------------------
__global__ __launch_bounds__(32) void moe_init_kernel(int* counts) {
    if (threadIdx.x < NEXP) counts[threadIdx.x] = 0;
}

// ---------------------------------------------------------------------------
// Gating: logits -> softmax -> top-2 -> per-expert token lists (+ x -> bf16)
__global__ __launch_bounds__(256) void moe_gate_kernel(
    const float* __restrict__ x, const float* __restrict__ gw,
    ushort_t* __restrict__ xb, int* __restrict__ counts,
    int* __restrict__ tokidx, float* __restrict__ tokw) {
    __shared__ float sG[NEXP * HDIM];              // 32 KB
    int tid = threadIdx.x;
    for (int i = tid; i < NEXP * HDIM; i += 256) sG[i] = gw[i];
    __syncthreads();

    int t = blockIdx.x * 256 + tid;
    const float* xr = x + (size_t)t * HDIM;
    float acc[NEXP];
#pragma unroll
    for (int e = 0; e < NEXP; ++e) acc[e] = 0.f;

    for (int h = 0; h < HDIM; ++h) {
        float xv = xr[h];
        xb[(size_t)t * HDIM + h] = f2bfu(xv);      // bf16 activation cache
#pragma unroll
        for (int e = 0; e < NEXP; ++e) acc[e] += xv * sG[e * HDIM + h];
    }

    float mx = acc[0];
#pragma unroll
    for (int e = 1; e < NEXP; ++e) mx = fmaxf(mx, acc[e]);
    float p[NEXP], sum = 0.f;
#pragma unroll
    for (int e = 0; e < NEXP; ++e) { p[e] = __expf(acc[e] - mx); sum += p[e]; }
    float inv = __builtin_amdgcn_rcpf(sum);

    int b0 = 0; float v0 = p[0];
#pragma unroll
    for (int e = 1; e < NEXP; ++e) if (p[e] > v0) { v0 = p[e]; b0 = e; }
    int b1 = (b0 == 0) ? 1 : 0; float v1 = p[b1];
#pragma unroll
    for (int e = 0; e < NEXP; ++e)
        if (e != b0 && p[e] > v1) { v1 = p[e]; b1 = e; }

    int pos0 = atomicAdd(&counts[b0], 1);
    tokidx[b0 * TOK + pos0] = t; tokw[b0 * TOK + pos0] = v0 * inv;
    int pos1 = atomicAdd(&counts[b1], 1);
    tokidx[b1 * TOK + pos1] = t; tokw[b1 * TOK + pos1] = v1 * inv;
}

// ---------------------------------------------------------------------------
// SwiGLU FFN over a 32-token tile (2 WMMA M-tiles sharing each B fragment):
//   y = (silu(X Wg^T) * (X Wu^T)) Wd^T * w
template <int IDIM, bool ROUTED>
__global__ __launch_bounds__(256) void moe_ffn_kernel(
    const ushort_t* __restrict__ xb,
    const float* __restrict__ Wg, const float* __restrict__ Wu,
    const float* __restrict__ Wd,
    const int* __restrict__ counts, const int* __restrict__ tokidx,
    const float* __restrict__ tokw, float* __restrict__ out) {
    constexpr int CH  = 256;             // I-chunk held in LDS
    constexpr int NCH = IDIM / CH;       // 8 routed, 4 shared
    constexpr int NTA = CH / (16 * 8);   // 2 N-tiles/wave, phase A
    constexpr int NTB = HDIM / (16 * 8); // 4 N-tiles/wave, phase B

    __shared__ __align__(16) ushort_t sX[MTILE * HDIM];  // 32 KB
    __shared__ __align__(16) ushort_t sI[MTILE * CH];    // 16 KB
    __shared__ float sWt[MTILE];
    __shared__ int   sTk[MTILE];

    int tile, cnt = TOK, eidx = 0;
    const float *wg = Wg, *wu = Wu, *wd = Wd;
    if (ROUTED) {
        eidx = blockIdx.x / MAXT;
        tile = blockIdx.x % MAXT;
        cnt  = counts[eidx];
        if (tile * MTILE >= cnt) return;             // uniform early exit
        wg = Wg + (size_t)eidx * IDIM * HDIM;
        wu = Wu + (size_t)eidx * IDIM * HDIM;
        wd = Wd + (size_t)eidx * HDIM * IDIM;
    } else {
        tile = blockIdx.x;
    }

    int tid = threadIdx.x;
    if (tid < MTILE) {
        int mg = tile * MTILE + tid;
        int t; float w;
        if (ROUTED) {
            if (mg < cnt) { t = tokidx[eidx * TOK + mg];
                            w = tokw[eidx * TOK + mg]; }
            else          { t = tokidx[eidx * TOK]; w = 0.f; }
        } else { t = mg; w = 1.f; }
        sTk[tid] = t; sWt[tid] = w;
    }
    __syncthreads();

    // stage X tile (32 x 512 bf16) into LDS via async global->LDS B128
    // (tracked by ASYNCcnt; LDS byte offset = low 32 bits of generic addr)
    {
        unsigned sXoff = (unsigned)(unsigned long long)(void*)&sX[0];
        for (int idx = tid; idx < MTILE * HDIM / 8; idx += 256) {
            int m = idx >> 6, off = (idx & 63) * 8;
            const ushort_t* src = xb + (size_t)sTk[m] * HDIM + off;
            unsigned dst = sXoff + idx * 16;
            asm volatile("global_load_async_to_lds_b128 %0, %1, off"
                         :: "v"(dst), "v"(src) : "memory");
        }
        asm volatile("s_wait_asynccnt 0" ::: "memory");
    }
    __syncthreads();

    const int lane = tid & 31, wv = tid >> 5;
    const int half = lane >> 4, l15 = lane & 15;

    v8f cacc[NTB][2];
#pragma unroll
    for (int nt = 0; nt < NTB; ++nt) {
        cacc[nt][0] = (v8f){};
        cacc[nt][1] = (v8f){};
    }

    for (int c = 0; c < NCH; ++c) {
        const int colbase = c * CH;
        // ---- Phase A: inter chunk = silu(X Wg^T) * (X Wu^T) -> sI ----
        for (int nt = 0; nt < NTA; ++nt) {
            int nb   = (wv * NTA + nt) * 16;           // within chunk
            int irow = colbase + nb + l15;             // row of Wg/Wu
            const float* pg = wg + (size_t)irow * HDIM + 8 * half;
            const float* pu = wu + (size_t)irow * HDIM + 8 * half;
            const ushort_t* pa0 = sX + l15 * HDIM + 8 * half;
            const ushort_t* pa1 = sX + (16 + l15) * HDIM + 8 * half;
            v8f cg0 = (v8f){}, cg1 = (v8f){}, cu0 = (v8f){}, cu1 = (v8f){};
            for (int k = 0; k < HDIM / 32; ++k) {
                __builtin_prefetch(pg + 32 * k + 128, 0, 1);
                __builtin_prefetch(pu + 32 * k + 128, 0, 1);
                v16bf bg = load_b16(pg + 32 * k);
                v16bf bu = load_b16(pu + 32 * k);
                v16bf a0 = load_a16(pa0 + 32 * k);
                v16bf a1 = load_a16(pa1 + 32 * k);
                cg0 = wmma_bf16(a0, bg, cg0);
                cg1 = wmma_bf16(a1, bg, cg1);
                cu0 = wmma_bf16(a0, bu, cu0);
                cu1 = wmma_bf16(a1, bu, cu1);
            }
#pragma unroll
            for (int mt = 0; mt < 2; ++mt) {
                v8f g = mt ? cg1 : cg0;
                v8f u = mt ? cu1 : cu0;
#pragma unroll
                for (int j = 0; j < 8; ++j) {
                    float gg = g[j];
                    float sig = __builtin_amdgcn_rcpf(1.f + __expf(-gg));
                    float s = gg * sig * u[j];
                    sI[(mt * 16 + 8 * half + j) * CH + nb + l15] = f2bfu(s);
                }
            }
        }
        __syncthreads();
        // ---- Phase B: partial y += inter_chunk @ Wd_chunk^T ----
        for (int nt = 0; nt < NTB; ++nt) {
            int nb = (wv * NTB + nt) * 16;             // h tile base
            const float* pd =
                wd + (size_t)(nb + l15) * IDIM + colbase + 8 * half;
            const ushort_t* pa0 = sI + l15 * CH + 8 * half;
            const ushort_t* pa1 = sI + (16 + l15) * CH + 8 * half;
            v8f c0 = cacc[nt][0], c1 = cacc[nt][1];
            for (int k = 0; k < CH / 32; ++k) {
                __builtin_prefetch(pd + 32 * k + 128, 0, 1);
                v16bf b  = load_b16(pd + 32 * k);
                v16bf a0 = load_a16(pa0 + 32 * k);
                v16bf a1 = load_a16(pa1 + 32 * k);
                c0 = wmma_bf16(a0, b, c0);
                c1 = wmma_bf16(a1, b, c1);
            }
            cacc[nt][0] = c0; cacc[nt][1] = c1;
        }
        __syncthreads();                               // protect sI rewrite
    }

    // ---- Epilogue: scale by gate weight, combine into out ----
#pragma unroll
    for (int nt = 0; nt < NTB; ++nt) {
        int h = (wv * NTB + nt) * 16 + l15;
#pragma unroll
        for (int mt = 0; mt < 2; ++mt) {
#pragma unroll
            for (int j = 0; j < 8; ++j) {
                int m = mt * 16 + 8 * half + j;
                float val = cacc[nt][mt][j] * sWt[m];
                float* dst = out + (size_t)sTk[m] * HDIM + h;
                if (ROUTED) atomicAdd(dst, val);
                else        *dst = val;
            }
        }
    }
}

// ---------------------------------------------------------------------------
extern "C" void kernel_launch(void* const* d_in, const int* in_sizes, int n_in,
                              void* d_out, int out_size, void* d_ws,
                              size_t ws_size, hipStream_t stream) {
    (void)in_sizes; (void)n_in; (void)out_size; (void)ws_size;
    const float* x       = (const float*)d_in[0];   // [1,2048,512]
    const float* gate_w  = (const float*)d_in[1];   // [16,512]
    const float* w_gate  = (const float*)d_in[2];   // [16,2048,512]
    const float* w_up    = (const float*)d_in[3];   // [16,2048,512]
    const float* w_down  = (const float*)d_in[4];   // [16,512,2048]
    const float* sw_gate = (const float*)d_in[5];   // [1024,512]
    const float* sw_up   = (const float*)d_in[6];   // [1024,512]
    const float* sw_down = (const float*)d_in[7];   // [512,1024]
    float* out = (float*)d_out;

    char* ws = (char*)d_ws;
    int*      counts = (int*)ws;                                 // 64 B
    int*      tokidx = (int*)(ws + 256);                         // E*T*4
    float*    tokw   = (float*)(ws + 256 + NEXP * TOK * 4);      // E*T*4
    ushort_t* xb     = (ushort_t*)(ws + 256 + NEXP * TOK * 8);   // T*H*2

    moe_init_kernel<<<1, 32, 0, stream>>>(counts);
    moe_gate_kernel<<<TOK / 256, 256, 0, stream>>>(x, gate_w, xb, counts,
                                                   tokidx, tokw);
    // shared expert first: plain stores initialize out
    moe_ffn_kernel<1024, false><<<TOK / MTILE, 256, 0, stream>>>(
        xb, sw_gate, sw_up, sw_down, nullptr, nullptr, nullptr, out);
    // routed experts: atomic accumulate
    moe_ffn_kernel<2048, true><<<NEXP * MAXT, 256, 0, stream>>>(
        xb, w_gate, w_up, w_down, counts, tokidx, tokw, out);
}
